// s2cnn_regression_75505525063916
// MI455X (gfx1250) — matrix operand order repaired and checked by reading, not compiled, not run on hardware
//
#include <hip/hip_runtime.h>
#include <math.h>

typedef __attribute__((ext_vector_type(16))) _Float16 v16h;
typedef __attribute__((ext_vector_type(8)))  float    v8f;

#define DPI 3.14159265358979323846

// ============================================================================
// Generic strided batched GEMM on WMMA (f32 in/out, f16 hi/lo split, f32 acc)
//   D[batch][row][col] (+)= sign * sum_g sum_k A[batch,g][row,k]*B[batch,g][k,col]*sc[k,col%mod]
// Composite index support: off(i) = (i/div)*hi + (i%div)*lo for batch and cols.
// LDS layout: A tile [row][k] (16x32), B tile [col][k] (16x32, transposed) so
// every lane's fragment is two contiguous 16B chunks -> ds_load_b128 pairs.
// ============================================================================
struct GP {
  int M, N, Kd, G, batch;
  const float* A;  int aG, aBdiv, aBhi, aBlo, aRow, aK;
  const float* Bm; int bG, bBdiv, bBhi, bBlo, bK, bCdiv, bChi, bClo;
  const float* sc; int sK, sMod;
  float* D;        int dBdiv, dBhi, dBlo, dRow, dCdiv, dChi, dClo;
  int acc; float sign; int relu;
};

__global__ __launch_bounds__(32) void k_gemm(GP p) {
  const int tilesN = (p.N + 15) >> 4;
  const int tm = ((int)blockIdx.x / tilesN) * 16;
  const int tn = ((int)blockIdx.x % tilesN) * 16;
  const int bb = blockIdx.y;
  const int lane = threadIdx.x;
  const float* Ab = p.A  + (bb / p.aBdiv) * p.aBhi + (bb % p.aBdiv) * p.aBlo;
  const float* Bb = p.Bm + (bb / p.bBdiv) * p.bBhi + (bb % p.bBdiv) * p.bBlo;
  __shared__ _Float16 Ah[512], Al[512], Bh[512], Bl[512];
  v8f acc = {0.f,0.f,0.f,0.f,0.f,0.f,0.f,0.f};
  for (int g = 0; g < p.G; ++g) {
    const float* Ag = Ab + (size_t)g * p.aG;
    const float* Bg = Bb + (size_t)g * p.bG;
    for (int k0 = 0; k0 < p.Kd; k0 += 32) {
      for (int it = 0; it < 16; ++it) {
        int idx = it * 32 + lane;
        { // A tile element (row r, k kk): stored [r][kk]
          int r = idx >> 5, kk = idx & 31;
          float v = 0.f;
          int row = tm + r, kq = k0 + kk;
          if (row < p.M && kq < p.Kd) v = Ag[(size_t)row * p.aRow + (size_t)kq * p.aK];
          _Float16 hv = (_Float16)v;
          Ah[idx] = hv; Al[idx] = (_Float16)(v - (float)hv);
        }
        { // B tile element (k r2, col c2): stored TRANSPOSED [c2][r2]
          int r2 = idx >> 4, c2 = idx & 15;
          float v = 0.f;
          int kq = k0 + r2, col = tn + c2;
          if (kq < p.Kd && col < p.N) {
            int coff = (col / p.bCdiv) * p.bChi + (col % p.bCdiv) * p.bClo;
            v = Bg[(size_t)kq * p.bK + coff];
            if (p.sc) v *= p.sc[(size_t)kq * p.sK + (col % p.sMod)];
          }
          _Float16 hv = (_Float16)v;
          Bh[c2*32 + r2] = hv; Bl[c2*32 + r2] = (_Float16)(v - (float)hv);
        }
      }
      // prefetch next A K-tile (lowers to global_prefetch_b8)
      if (k0 + 32 < p.Kd) {
        int prow = tm + (lane & 15);
        if (prow < p.M)
          __builtin_prefetch((const void*)(Ag + (size_t)prow * p.aRow + (size_t)(k0 + 32) * p.aK), 0, 1);
      }
      __syncthreads();
      // build fragments with k = 16*(lane>>4) + e for A and B alike (contiguous 16B LDS reads)
      v16h ah, al, bh, bl;
      const int mrow  = lane & 15;
      const int abase = (lane >> 4) * 16;
      #pragma unroll
      for (int e = 0; e < 8; ++e) {
        ah[e]     = Ah[mrow*32 + abase + e];
        ah[e + 8] = Ah[mrow*32 + abase + 8 + e];
        al[e]     = Al[mrow*32 + abase + e];
        al[e + 8] = Al[mrow*32 + abase + 8 + e];
      }
      const int ncol  = lane & 15;
      const int bbase = (lane >> 4) * 16;
      #pragma unroll
      for (int e = 0; e < 16; ++e) {
        bh[e] = Bh[ncol*32 + bbase + e];
        bl[e] = Bl[ncol*32 + bbase + e];
      }
      // f32-accurate product via hi/lo split: 3 WMMAs
      acc = __builtin_amdgcn_wmma_f32_16x16x32_f16(false, ah, false, bh, (short)0, acc, false, false);
      acc = __builtin_amdgcn_wmma_f32_16x16x32_f16(false, ah, false, bl, (short)0, acc, false, false);
      acc = __builtin_amdgcn_wmma_f32_16x16x32_f16(false, al, false, bh, (short)0, acc, false, false);
      __syncthreads();
    }
  }
  float* Db = p.D + (bb / p.dBdiv) * p.dBhi + (bb % p.dBdiv) * p.dBlo;
  const int ncol = lane & 15;
  const int col = tn + ncol;
  if (col < p.N) {
    int coff = (col / p.dCdiv) * p.dChi + (col % p.dCdiv) * p.dClo;
    const int rbase = (lane >> 4) * 8;
    #pragma unroll
    for (int r = 0; r < 8; ++r) {
      int row = tm + rbase + r;
      if (row < p.M) {
        float v = acc[r] * p.sign;
        float* dp = Db + (size_t)row * p.dRow + coff;
        if (p.acc) v += *dp;
        if (p.relu && v < 0.f) v = 0.f;
        *dp = v;
      }
    }
  }
}

// ============================================================================
// Table-generation kernels (VALU; run once per call, unmeasured)
// ============================================================================

// quadrature weights w_k for SOFT grid of bandwidth b
__global__ void k_quad(float* qw, int b) {
  int k = blockIdx.x * blockDim.x + threadIdx.x;
  if (k >= 2*b) return;
  double beta = DPI * (2*k + 1) / (4.0*b);
  double s = 0.0;
  for (int j = 0; j < b; ++j)
    s += sin(DPI * (2*k + 1) * (2*j + 1) / (4.0*b)) / (2*j + 1);
  qw[k] = (float)((2.0/b) * sin(beta) * s);
}

// 8 planes: fwd_re, fwd_im, fwdT_re, fwdT_im, inv_re, inv_im, invT_re, invT_im
// Ffwd[j][t] = exp(-2*pi*i*(j-K/2)*t/K)  (fft + fftshift folded)
// Finv[t][j] = exp(+2*pi*i*(j-K/2)*t/K)  (ifftshift + K*ifft folded)
__global__ void k_dft(float* d, int K) {
  int tot = K*K;
  for (int idx = blockIdx.x*blockDim.x + threadIdx.x; idx < tot; idx += gridDim.x*blockDim.x) {
    int j = idx / K, t = idx % K;
    double af = -2.0*DPI*(double)(j - K/2)*(double)t/(double)K;
    float cr = (float)cos(af), ci = (float)sin(af);
    d[0*tot + j*K + t] = cr;   d[1*tot + j*K + t] = ci;
    d[2*tot + t*K + j] = cr;   d[3*tot + t*K + j] = ci;
    d[4*tot + t*K + j] = cr;   d[5*tot + t*K + j] = -ci;
    d[6*tot + j*K + t] = cr;   d[7*tot + j*K + t] = -ci;
  }
}

// Wigner-d explicit sum (Jacobi form), log-factorials in LDS
__device__ double wigd(const double* lf, int l, int mp, int m, double cb2, double sb2) {
  if (mp < -l || mp > l || m < -l || m > l) return 0.0;
  if (l == 0) return 1.0;
  int smin = m - mp; if (smin < 0) smin = 0;
  int smax = l + m;  if (l - mp < smax) smax = l - mp;
  double lc = log(cb2), ls = log(sb2);
  double pref = 0.5 * (lf[l+m] + lf[l-m] + lf[l+mp] + lf[l-mp]);
  double sum = 0.0;
  for (int s = smin; s <= smax; ++s) {
    double lt = pref - lf[l+m-s] - lf[s] - lf[mp-m+s] - lf[l-mp-s]
              + (double)(2*l + m - mp - 2*s) * lc + (double)(mp - m + 2*s) * ls;
    double t = exp(lt);
    sum += ((mp - m + s) & 1) ? -t : t;
  }
  return sum;
}

#define LFACT_INIT() \
  __shared__ double lf[70]; \
  if (threadIdx.x == 0) { lf[0] = 0.0; for (int i_ = 1; i_ < 70; ++i_) lf[i_] = lf[i_-1] + log((double)i_); } \
  __syncthreads()

// Dpad[k][l][mi][ni] = d^l_{mi-b, ni-b}(beta_k), DW = Dpad * qw[k], zero-padded
__global__ void k_wigner_grid(float* dpad, float* dw, const float* qw, int b) {
  LFACT_INIT();
  int K = 2*b, L = b;
  long tot = (long)K * L * K * K;
  for (long idx = blockIdx.x*(long)blockDim.x + threadIdx.x; idx < tot; idx += (long)gridDim.x*blockDim.x) {
    int ni = (int)(idx % K);
    int mi = (int)((idx / K) % K);
    int l  = (int)((idx / ((long)K*K)) % L);
    int k  = (int)( idx / ((long)K*K*L));
    double beta = DPI * (2*k + 1) / (4.0*b);
    double v = wigd(lf, l, mi - b, ni - b, cos(beta*0.5), sin(beta*0.5));
    dpad[idx] = (float)v;
    dw[idx]   = (float)(v * (double)qw[k]);
  }
}

// d0pad[l][mi][ni] = d^l_{mi-K/2, ni-K/2}(beta0)
__global__ void k_wigner_b0(float* dst, double beta0, int L, int K) {
  LFACT_INIT();
  long tot = (long)L * K * K;
  double c = cos(beta0*0.5), s = sin(beta0*0.5);
  for (long idx = blockIdx.x*(long)blockDim.x + threadIdx.x; idx < tot; idx += (long)gridDim.x*blockDim.x) {
    int ni = (int)(idx % K);
    int mi = (int)((idx / K) % K);
    int l  = (int)( idx / ((long)K*K));
    dst[idx] = (float)wigd(lf, l, mi - K/2, ni - K/2, c, s);
  }
}

// D0W[k][l][mi] = qw[k]*d^l_{mi-b,0}(beta_k)   (s2 projection table)
__global__ void k_wigner_s2grid(float* d0w, const float* qw, int b) {
  LFACT_INIT();
  int K = 2*b, L = b;
  long tot = (long)K * L * K;
  for (long idx = blockIdx.x*(long)blockDim.x + threadIdx.x; idx < tot; idx += (long)gridDim.x*blockDim.x) {
    int mi = (int)(idx % K);
    int l  = (int)((idx / K) % L);
    int k  = (int)( idx / ((long)K*L));
    double beta = DPI * (2*k + 1) / (4.0*b);
    d0w[idx] = (float)(wigd(lf, l, mi - b, 0, cos(beta*0.5), sin(beta*0.5)) * (double)qw[k]);
  }
}

// d0s2[l][ni] = d^l_{ni-K/2,0}(beta0)
__global__ void k_wigner_s2b0(float* dst, double beta0, int L, int K) {
  LFACT_INIT();
  long tot = (long)L * K;
  double c = cos(beta0*0.5), s = sin(beta0*0.5);
  for (long idx = blockIdx.x*(long)blockDim.x + threadIdx.x; idx < tot; idx += (long)gridDim.x*blockDim.x) {
    int ni = (int)(idx % K);
    int l  = (int)(idx / K);
    dst[idx] = (float)wigd(lf, l, ni - K/2, 0, c, s);
  }
}

// so3 weight spectrum: S[c][pi][f][ni] = sum_{a,g} w[c,f,a*6+g] e^{i(m alpha_a + n gamma_g)}
__global__ void k_make_S(const float* w, float* Sre, float* Sim, int C, int F, int bo) {
  int K = 2*bo;
  long tot = (long)C * K * F * K;
  for (long idx = blockIdx.x*(long)blockDim.x + threadIdx.x; idx < tot; idx += (long)gridDim.x*blockDim.x) {
    int ni = (int)(idx % K);
    int f  = (int)((idx / K) % F);
    int pi = (int)((idx / ((long)K*F)) % K);
    int c  = (int)( idx / ((long)K*F*K));
    int m = pi - bo, n = ni - bo;
    float re = 0.f, im = 0.f;
    if (m > -bo && n > -bo) {
      const float* wp = w + ((size_t)c*F + f) * 36;
      for (int a = 0; a < 6; ++a)
        for (int g2 = 0; g2 < 6; ++g2) {
          double ang = (2.0*DPI/6.0) * (double)(m*a + n*g2);
          float ww = wp[a*6 + g2];
          re += ww * (float)cos(ang);
          im += ww * (float)sin(ang);
        }
    }
    Sre[idx] = re; Sim[idx] = im;
  }
}

// s2 weight spectrum: S2[c][f][ni] = sum_j w[c,f,j] e^{-i n alpha_j}
__global__ void k_make_S2(const float* w, float* Sre, float* Sim, int C, int F, int K) {
  long tot = (long)C * F * K;
  for (long idx = blockIdx.x*(long)blockDim.x + threadIdx.x; idx < tot; idx += (long)gridDim.x*blockDim.x) {
    int ni = (int)(idx % K);
    int f  = (int)((idx / K) % F);
    int c  = (int)( idx / ((long)K*F));
    int n = ni - K/2;
    float re = 0.f, im = 0.f;
    if (n > -K/2) {
      const float* wp = w + ((size_t)c*F + f) * 6;
      for (int j = 0; j < 6; ++j) {
        double ang = -(2.0*DPI/6.0) * (double)(n*j);
        re += wp[j] * (float)cos(ang);
        im += wp[j] * (float)sin(ang);
      }
    }
    Sre[idx] = re; Sim[idx] = im;
  }
}

// so3_integrate (b=2) + linear head. h [4][256][4][4][4] -> out[4]
__global__ void k_integrate(const float* h, const float* qw2, const float* lin_w,
                            const float* lin_b, float* out) {
  int b = blockIdx.x;
  int f = threadIdx.x;           // 256 threads
  const float* hp = h + ((size_t)b*256 + f) * 64;
  float s = 0.f;
  for (int k = 0; k < 4; ++k) {
    float w = qw2[k];
    for (int ag = 0; ag < 16; ++ag) s += hp[k*16 + ag] * w;
  }
  s *= (1.f/16.f);
  __shared__ float red[256];
  red[f] = s * lin_w[f];
  __syncthreads();
  for (int st2 = 128; st2 > 0; st2 >>= 1) { if (f < st2) red[f] += red[f + st2]; __syncthreads(); }
  if (f == 0) out[b] = red[0] + lin_b[0];
}

// ============================================================================
// Host orchestration
// ============================================================================
struct Ctx {
  hipStream_t st;
  float* dft[5];   // K = 64,32,16,8,4
  float* qw[5];
  float* dpad[5];
  float* dw[5];
  float *C0r, *C0i, *C1r, *C1i, *XHr, *XHi, *YPr, *YPi;
};
static inline int kidx(int K) { return K==64?0 : K==32?1 : K==16?2 : K==8?3 : 4; }

static void G(hipStream_t st, int M, int N, int Kd, int Gg, int batch,
              const float* A, int aG, int aBdiv, int aBhi, int aBlo, int aRow, int aK,
              const float* B, int bG, int bBdiv, int bBhi, int bBlo, int bK, int bCdiv, int bChi, int bClo,
              const float* sc, int sK, int sMod,
              float* D, int dBdiv, int dBhi, int dBlo, int dRow, int dCdiv, int dChi, int dClo,
              int acc, float sign, int relu) {
  GP p;
  p.M=M; p.N=N; p.Kd=Kd; p.G=Gg; p.batch=batch;
  p.A=A; p.aG=aG; p.aBdiv=aBdiv; p.aBhi=aBhi; p.aBlo=aBlo; p.aRow=aRow; p.aK=aK;
  p.Bm=B; p.bG=bG; p.bBdiv=bBdiv; p.bBhi=bBhi; p.bBlo=bBlo; p.bK=bK; p.bCdiv=bCdiv; p.bChi=bChi; p.bClo=bClo;
  p.sc=sc; p.sK=sK; p.sMod=sMod;
  p.D=D; p.dBdiv=dBdiv; p.dBhi=dBhi; p.dBlo=dBlo; p.dRow=dRow; p.dCdiv=dCdiv; p.dChi=dChi; p.dClo=dClo;
  p.acc=acc; p.sign=sign; p.relu=relu;
  dim3 grid((unsigned)(((M+15)/16) * ((N+15)/16)), (unsigned)batch, 1);
  k_gemm<<<grid, dim3(32,1,1), 0, st>>>(p);
}

// synthesis (YP -> C0) + inverse FFT2 (C0 -> C1 -> out)
static void synth_inv(Ctx& cx, int Bn, int F, int bo, float* out, int accOut, int reluOut) {
  hipStream_t st = cx.st;
  int Ko = 2*bo;
  const float* Dp = cx.dpad[kidx(Ko)];
  float* Do = cx.dft[kidx(Ko)];
  for (int t = 0; t < 2; ++t) {
    const float* Bp = t ? cx.YPi : cx.YPr;
    float* Dd = t ? cx.C0i : cx.C0r;
    G(st, Ko, Bn*F, bo, 1, Ko*Ko,
      Dp, 0, 1,1,0, bo*Ko*Ko, Ko*Ko,
      Bp, 0, Ko, F*Ko, 1, Bn*Ko*F*Ko, F, Ko*F*Ko, Ko,
      nullptr, 0, 1,
      Dd, 1,1,0, Ko*Ko, 1, Ko*Ko*Ko, 0, 0, 1.f, 0);
  }
  const float* i_re = Do + 4*Ko*Ko; const float* i_im = Do + 5*Ko*Ko;
  const float* t_re = Do + 6*Ko*Ko; const float* t_im = Do + 7*Ko*Ko;
  int q = Bn*F*Ko;
  G(st, Ko,Ko,Ko,1,q, i_re,0,1,0,0,Ko,1, cx.C0r,0, 1,Ko*Ko,0, Ko,1,1,0, nullptr,0,1, cx.C1r, 1,Ko*Ko,0, Ko,1,1,0, 0,  1.f, 0);
  G(st, Ko,Ko,Ko,1,q, i_im,0,1,0,0,Ko,1, cx.C0i,0, 1,Ko*Ko,0, Ko,1,1,0, nullptr,0,1, cx.C1r, 1,Ko*Ko,0, Ko,1,1,0, 1, -1.f, 0);
  G(st, Ko,Ko,Ko,1,q, i_re,0,1,0,0,Ko,1, cx.C0i,0, 1,Ko*Ko,0, Ko,1,1,0, nullptr,0,1, cx.C1i, 1,Ko*Ko,0, Ko,1,1,0, 0,  1.f, 0);
  G(st, Ko,Ko,Ko,1,q, i_im,0,1,0,0,Ko,1, cx.C0r,0, 1,Ko*Ko,0, Ko,1,1,0, nullptr,0,1, cx.C1i, 1,Ko*Ko,0, Ko,1,1,0, 1,  1.f, 0);
  int rows = Bn*F*Ko*Ko;
  G(st, rows, Ko, Ko, 1, 1, cx.C1r,0,1,0,0,Ko,1, t_re,0,1,0,0,Ko,1,1,0, nullptr,0,1, out, 1,0,0, Ko,1,1,0, accOut, 1.f, 0);
  G(st, rows, Ko, Ko, 1, 1, cx.C1i,0,1,0,0,Ko,1, t_im,0,1,0,0,Ko,1,1,0, nullptr,0,1, out, 1,0,0, Ko,1,1,0, 1, -1.f, reluOut);
}

struct CMB { const float* A; const float* Bp; float* Dp; int acc; float sg; };

static void run_so3(Ctx& cx, const float* in, float* out, int Bn, int C, int F, int bi, int bo,
                    const float* Sre, const float* Sim, const float* d0,
                    int accOut, int reluOut) {
  hipStream_t st = cx.st;
  int Ki = 2*bi, Ko = 2*bo;
  float* Di = cx.dft[kidx(Ki)];
  const float* f_re = Di; const float* f_im = Di + Ki*Ki;
  const float* fT_re = Di + 2*Ki*Ki; const float* fT_im = Di + 3*Ki*Ki;
  // forward FFT, last axis (gamma)
  int rows = Bn*C*Ki*Ki;
  G(st, rows, Ki, Ki, 1, 1, in,0,1,0,0,Ki,1, fT_re,0,1,0,0,Ki,1,1,0, nullptr,0,1, cx.C1r, 1,0,0, Ki,1,1,0, 0, 1.f, 0);
  G(st, rows, Ki, Ki, 1, 1, in,0,1,0,0,Ki,1, fT_im,0,1,0,0,Ki,1,1,0, nullptr,0,1, cx.C1i, 1,0,0, Ki,1,1,0, 0, 1.f, 0);
  // forward FFT, alpha axis: C0 = Ffwd * C1 (complex)
  int q2 = Bn*C*Ki;
  G(st, Ki,Ki,Ki,1,q2, f_re,0,1,0,0,Ki,1, cx.C1r,0, 1,Ki*Ki,0, Ki,1,1,0, nullptr,0,1, cx.C0r, 1,Ki*Ki,0, Ki,1,1,0, 0,  1.f, 0);
  G(st, Ki,Ki,Ki,1,q2, f_im,0,1,0,0,Ki,1, cx.C1i,0, 1,Ki*Ki,0, Ki,1,1,0, nullptr,0,1, cx.C0r, 1,Ki*Ki,0, Ki,1,1,0, 1, -1.f, 0);
  G(st, Ki,Ki,Ki,1,q2, f_re,0,1,0,0,Ki,1, cx.C1i,0, 1,Ki*Ki,0, Ki,1,1,0, nullptr,0,1, cx.C0i, 1,Ki*Ki,0, Ki,1,1,0, 0,  1.f, 0);
  G(st, Ki,Ki,Ki,1,q2, f_im,0,1,0,0,Ki,1, cx.C1r,0, 1,Ki*Ki,0, Ki,1,1,0, nullptr,0,1, cx.C0i, 1,Ki*Ki,0, Ki,1,1,0, 1,  1.f, 0);
  // xh[l][b][c][mi][ni] = sum_k qw[k] d^l(beta_k) X
  const float* DWt = cx.dw[kidx(Ki)];
  G(st, bo, Bn*C, Ki, 1, Ki*Ki,
    DWt, 0, 1,1,0, Ki*Ki, bi*Ki*Ki,
    cx.C0r, 0, 1,1,0, Ki*Ki, 1, Ki*Ki*Ki, 0, nullptr,0,1,
    cx.XHr, 1,1,0, Bn*C*Ki*Ki, 1, Ki*Ki, 0, 0, 1.f, 0);
  G(st, bo, Bn*C, Ki, 1, Ki*Ki,
    DWt, 0, 1,1,0, Ki*Ki, bi*Ki*Ki,
    cx.C0i, 0, 1,1,0, Ki*Ki, 1, Ki*Ki*Ki, 0, nullptr,0,1,
    cx.XHi, 1,1,0, Bn*C*Ki*Ki, 1, Ki*Ki, 0, 0, 1.f, 0);
  // channel mix per degree l: YP[l][b][mi][f][ni] = sum_{c,p} xh * S * d0_l
  int sh = bi - bo;
  for (int l = 0; l < bo; ++l) {
    const float* Ar = cx.XHr + (size_t)l*Bn*C*Ki*Ki + sh*Ki + sh;
    const float* Ai = cx.XHi + (size_t)l*Bn*C*Ki*Ki + sh*Ki + sh;
    const float* scl = d0 + (size_t)l*Ko*Ko;
    float* Dr = cx.YPr + (size_t)l*Bn*Ko*F*Ko;
    float* Dq = cx.YPi + (size_t)l*Bn*Ko*F*Ko;
    CMB cmb[4] = { {Ar, Sre, Dr, 0, 1.f}, {Ai, Sim, Dr, 1, -1.f},
                   {Ar, Sim, Dq, 0, 1.f}, {Ai, Sre, Dq, 1, 1.f} };
    for (int t = 0; t < 4; ++t)
      G(st, Ko, F*Ko, Ko, C, Bn,
        cmb[t].A, Ki*Ki, 1, C*Ki*Ki, 0, Ki, 1,
        cmb[t].Bp, Ko*F*Ko, 1,0,0, F*Ko, 1,1,0,
        scl, Ko, Ko,
        cmb[t].Dp, 1, Ko*F*Ko, 0, F*Ko, 1,1,0, cmb[t].acc, cmb[t].sg, 0);
  }
  synth_inv(cx, Bn, F, bo, out, accOut, reluOut);
}

static void run_s2(Ctx& cx, const float* x, float* out, int Bn, int C, int F, int b,
                   const float* S2r, const float* S2i, const float* d0w, const float* d0v) {
  hipStream_t st = cx.st;
  int K = 2*b;
  float* Di = cx.dft[kidx(K)];
  const float* fT_re = Di + 2*K*K; const float* fT_im = Di + 3*K*K;
  int rows = Bn*C*K;
  G(st, rows, K, K, 1, 1, x,0,1,0,0,K,1, fT_re,0,1,0,0,K,1,1,0, nullptr,0,1, cx.C0r, 1,0,0, K,1,1,0, 0, 1.f, 0);
  G(st, rows, K, K, 1, 1, x,0,1,0,0,K,1, fT_im,0,1,0,0,K,1,1,0, nullptr,0,1, cx.C0i, 1,0,0, K,1,1,0, 0, 1.f, 0);
  // xh[l][b][c][mi] = sum_k D0W[k][l][mi] * X[b][c][k][mi]
  G(st, b, Bn*C, K, 1, K,
    d0w, 0, 1,1,0, K, b*K,
    cx.C0r, 0, 1,1,0, K, 1, K*K, 0, nullptr,0,1,
    cx.XHr, 1,1,0, Bn*C*K, 1, K, 0, 0, 1.f, 0);
  G(st, b, Bn*C, K, 1, K,
    d0w, 0, 1,1,0, K, b*K,
    cx.C0i, 0, 1,1,0, K, 1, K*K, 0, nullptr,0,1,
    cx.XHi, 1,1,0, Bn*C*K, 1, K, 0, 0, 1.f, 0);
  // rank-1 per degree: YP[l][b][mi][f][ni] = sum_c xh[l,b,c,mi]*S2[c,f,ni]*d0_l[ni]
  for (int l = 0; l < b; ++l) {
    const float* Ar = cx.XHr + (size_t)l*Bn*C*K;
    const float* Ai = cx.XHi + (size_t)l*Bn*C*K;
    const float* scl = d0v + (size_t)l*K;
    float* Dr = cx.YPr + (size_t)l*Bn*K*F*K;
    float* Dq = cx.YPi + (size_t)l*Bn*K*F*K;
    CMB cmb[4] = { {Ar, S2r, Dr, 0, 1.f}, {Ai, S2i, Dr, 1, -1.f},
                   {Ar, S2i, Dq, 0, 1.f}, {Ai, S2r, Dq, 1, 1.f} };
    for (int t = 0; t < 4; ++t)
      G(st, K, F*K, C, 1, Bn,
        cmb[t].A, 0, 1, C*K, 0, 1, K,
        cmb[t].Bp, 0, 1,0,0, F*K, 1,1,0,
        scl, 0, K,
        cmb[t].Dp, 1, K*F*K, 0, F*K, 1,1,0, cmb[t].acc, cmb[t].sg, 0);
  }
  synth_inv(cx, Bn, F, b, out, 0, 1);
}

extern "C" void kernel_launch(void* const* d_in, const int* in_sizes, int n_in,
                              void* d_out, int out_size, void* d_ws, size_t ws_size,
                              hipStream_t stream) {
  (void)in_sizes; (void)n_in; (void)out_size; (void)ws_size;
  float* W = (float*)d_ws;
  size_t off = 0;
  auto alloc = [&](size_t n) { float* p = W + off; off += n; return p; };

  static const int bs[5] = {32, 16, 8, 4, 2};
  float* qw[5]; float* dftp[5]; float* dpad[5]; float* dw[5];
  for (int i = 0; i < 5; ++i) qw[i] = alloc(2*bs[i]);
  for (int i = 0; i < 5; ++i) { int K = 2*bs[i]; dftp[i] = alloc((size_t)8*K*K); }
  for (int i = 0; i < 5; ++i) {
    int b = bs[i]; size_t n = (size_t)(2*b)*b*(2*b)*(2*b);
    dpad[i] = alloc(n); dw[i] = alloc(n);
  }
  float* d0w_s2 = alloc((size_t)64*32*64);
  float* d0s2   = alloc((size_t)32*64);
  float* d0_32_16 = alloc((size_t)32*64*64);
  float* d0_16_16 = alloc((size_t)16*32*32);
  float* d0_16_8  = alloc((size_t)16*32*32);
  float* d0_8_8   = alloc((size_t)8*16*16);
  float* d0_8_4   = alloc((size_t)8*16*16);
  float* d0_4_4   = alloc((size_t)4*8*8);
  float* d0_4_2   = alloc((size_t)4*8*8);
  float* d0_2_2   = alloc((size_t)2*4*4);

  struct ConvDef { int C, F, bo; };
  static const ConvDef CD[12] = {
    {16,16,32},{16,32,16},{16,32,16},
    {32,32,16},{32,64,8},{32,64,8},
    {64,64,8},{64,128,4},{64,128,4},
    {128,128,4},{128,256,2},{128,256,2}};
  float* Sr[12]; float* Si[12];
  for (int i = 0; i < 12; ++i) {
    size_t n = (size_t)CD[i].C * (2*CD[i].bo) * CD[i].F * (2*CD[i].bo);
    Sr[i] = alloc(n); Si[i] = alloc(n);
  }
  float* S2r = alloc((size_t)4*16*64); float* S2i = alloc((size_t)4*16*64);

  const size_t SIGN = 16777216;      // 4*16*64^3
  float* SIG0 = alloc(SIGN); float* SIG1 = alloc(SIGN); float* SIG2 = alloc(SIGN);
  float* C0r = alloc(SIGN); float* C0i = alloc(SIGN);
  float* C1r = alloc(SIGN); float* C1i = alloc(SIGN);
  float* XHr = alloc(8388608); float* XHi = alloc(8388608);
  float* YPr = alloc(8388608); float* YPi = alloc(8388608);

  // ---------------- precompute tables ----------------
  for (int i = 0; i < 5; ++i) k_quad<<<1, 64, 0, stream>>>(qw[i], bs[i]);
  for (int i = 0; i < 5; ++i) {
    int K = 2*bs[i]; int tot = K*K;
    k_dft<<<(tot + 255)/256, 256, 0, stream>>>(dftp[i], K);
  }
  for (int i = 0; i < 5; ++i) {
    int b = bs[i]; long tot = (long)(2*b)*b*(2*b)*(2*b);
    k_wigner_grid<<<(unsigned)((tot + 255)/256), 256, 0, stream>>>(dpad[i], dw[i], qw[i], b);
  }
  k_wigner_s2grid<<<(64*32*64 + 255)/256, 256, 0, stream>>>(d0w_s2, qw[0], 32);
  k_wigner_s2b0<<<(32*64 + 255)/256, 256, 0, stream>>>(d0s2, DPI/16.0, 32, 64);
  k_wigner_b0<<<(32*64*64 + 255)/256, 256, 0, stream>>>(d0_32_16, DPI/16.0, 32, 64);
  k_wigner_b0<<<(16*32*32 + 255)/256, 256, 0, stream>>>(d0_16_16, DPI/16.0, 16, 32);
  k_wigner_b0<<<(16*32*32 + 255)/256, 256, 0, stream>>>(d0_16_8,  DPI/8.0,  16, 32);
  k_wigner_b0<<<(8*16*16 + 255)/256, 256, 0, stream>>>(d0_8_8,  DPI/8.0, 8, 16);
  k_wigner_b0<<<(8*16*16 + 255)/256, 256, 0, stream>>>(d0_8_4,  DPI/4.0, 8, 16);
  k_wigner_b0<<<(4*8*8 + 255)/256, 256, 0, stream>>>(d0_4_4, DPI/4.0, 4, 8);
  k_wigner_b0<<<(4*8*8 + 255)/256, 256, 0, stream>>>(d0_4_2, DPI/2.0, 4, 8);
  k_wigner_b0<<<(2*4*4 + 255)/256, 256, 0, stream>>>(d0_2_2, DPI/2.0, 2, 4);

  // ---------------- weight spectra ----------------
  const float* x = (const float*)d_in[0];
  k_make_S2<<<(4*16*64 + 255)/256, 256, 0, stream>>>((const float*)d_in[1], S2r, S2i, 4, 16, 64);
  for (int i = 0; i < 12; ++i) {
    long tot = (long)CD[i].C * (2*CD[i].bo) * CD[i].F * (2*CD[i].bo);
    k_make_S<<<(unsigned)((tot + 255)/256), 256, 0, stream>>>((const float*)d_in[2 + i], Sr[i], Si[i],
                                                              CD[i].C, CD[i].F, CD[i].bo);
  }

  // ---------------- network ----------------
  Ctx cx;
  cx.st = stream;
  for (int i = 0; i < 5; ++i) { cx.dft[i] = dftp[i]; cx.qw[i] = qw[i]; cx.dpad[i] = dpad[i]; cx.dw[i] = dw[i]; }
  cx.C0r = C0r; cx.C0i = C0i; cx.C1r = C1r; cx.C1i = C1i;
  cx.XHr = XHr; cx.XHi = XHi; cx.YPr = YPr; cx.YPi = YPi;

  run_s2(cx, x, SIG0, 4, 4, 16, 32, S2r, S2i, d0w_s2, d0s2);           // h0 = relu(s2_conv)
  // res block 1 (bw=32)
  run_so3(cx, SIG0, SIG1, 4, 16, 16, 32, 32, Sr[0], Si[0], d0_32_16, 0, 1);
  run_so3(cx, SIG1, SIG2, 4, 16, 32, 32, 16, Sr[1], Si[1], d0_16_16, 0, 0);
  run_so3(cx, SIG0, SIG2, 4, 16, 32, 32, 16, Sr[2], Si[2], d0_16_16, 1, 1);
  // res block 2 (bw=16)
  run_so3(cx, SIG2, SIG0, 4, 32, 32, 16, 16, Sr[3], Si[3], d0_16_8, 0, 1);
  run_so3(cx, SIG0, SIG1, 4, 32, 64, 16, 8,  Sr[4], Si[4], d0_8_8,  0, 0);
  run_so3(cx, SIG2, SIG1, 4, 32, 64, 16, 8,  Sr[5], Si[5], d0_8_8,  1, 1);
  // res block 3 (bw=8)
  run_so3(cx, SIG1, SIG0, 4, 64, 64, 8, 8,   Sr[6], Si[6], d0_8_4,  0, 1);
  run_so3(cx, SIG0, SIG2, 4, 64, 128, 8, 4,  Sr[7], Si[7], d0_4_4,  0, 0);
  run_so3(cx, SIG1, SIG2, 4, 64, 128, 8, 4,  Sr[8], Si[8], d0_4_4,  1, 1);
  // res block 4 (bw=4)
  run_so3(cx, SIG2, SIG0, 4, 128, 128, 4, 4, Sr[9], Si[9], d0_4_2,  0, 1);
  run_so3(cx, SIG0, SIG1, 4, 128, 256, 4, 2, Sr[10], Si[10], d0_2_2, 0, 0);
  run_so3(cx, SIG2, SIG1, 4, 128, 256, 4, 2, Sr[11], Si[11], d0_2_2, 1, 1);
  // integrate + linear head
  k_integrate<<<4, 256, 0, stream>>>(SIG1, qw[4], (const float*)d_in[14], (const float*)d_in[15],
                                     (float*)d_out);
}
